// TriUpdate_12094627906023
// MI455X (gfx1250) — matrix operand order repaired and checked by reading, not compiled, not run on hardware
//
#include <hip/hip_runtime.h>
#include <hip/hip_bf16.h>
#include <math.h>

typedef _Float16 v16h __attribute__((ext_vector_type(16)));
typedef float    v8f  __attribute__((ext_vector_type(8)));

#define M_TOK 65536          // 256*256 positions
#define TU_EPS 1e-5f

__device__ __forceinline__ float tu_sig(float x){ return 1.0f/(1.0f+__expf(-x)); }
__device__ __forceinline__ float tu_elu(float x){ return x>0.f ? x : (__expf(x)-1.f); }

// gfx1250 async copy: global -> LDS, 16 bytes per lane, tracked by ASYNCcnt.
// LDS byte offset = low 32 bits of the generic shared-pointer address.
__device__ __forceinline__ void tu_async_b128(unsigned lds_off, const float* gaddr){
  asm volatile("global_load_async_to_lds_b128 %0, %1, off"
               :: "v"(lds_off), "v"(gaddr) : "memory");
}
__device__ __forceinline__ void tu_wait_async(){
  asm volatile("s_wait_asynccnt 0x0" ::: "memory");
}

// ---------------------------------------------------------------- layernorm
// one wave per row; C = 128 or 64
__global__ void tu_ln_kernel(const float* __restrict__ x, const float* __restrict__ g,
                             const float* __restrict__ b, float* __restrict__ y, int C){
  int wave = threadIdx.x >> 5, lane = threadIdx.x & 31;
  int row  = blockIdx.x * (blockDim.x >> 5) + wave;
  const float* xr = x + (size_t)row * C;
  int per = C >> 5;
  float v[4];
  float s = 0.f;
  for(int e=0;e<per;e++){ v[e] = xr[lane + 32*e]; s += v[e]; }
  #pragma unroll
  for(int off=16;off>0;off>>=1) s += __shfl_xor(s, off, 32);
  float mean = s / (float)C;
  float q = 0.f;
  for(int e=0;e<per;e++){ float d = v[e]-mean; q += d*d; }
  #pragma unroll
  for(int off=16;off>0;off>>=1) q += __shfl_xor(q, off, 32);
  float rstd = rsqrtf(q / (float)C + TU_EPS);
  float* yr = y + (size_t)row * C;
  for(int e=0;e<per;e++){ int c = lane + 32*e; yr[c] = g[c]*(v[e]-mean)*rstd + b[c]; }
}

// ---------------------------------------------------------- instance norm
__global__ void tu_instat_kernel(const float* __restrict__ x, int C, float* __restrict__ stats){
  __shared__ float s1[256], s2[256];
  int c = blockIdx.x, t = threadIdx.x;
  float a = 0.f, b = 0.f;
  for(int m=t; m<M_TOK; m+=256){ float v = x[(size_t)m*C + c]; a += v; b += v*v; }
  s1[t]=a; s2[t]=b; __syncthreads();
  for(int off=128; off>0; off>>=1){
    if(t<off){ s1[t]+=s1[t+off]; s2[t]+=s2[t+off]; }
    __syncthreads();
  }
  if(t==0){
    float mean = s1[0]/(float)M_TOK;
    float var  = s2[0]/(float)M_TOK - mean*mean;
    stats[2*c] = mean; stats[2*c+1] = rsqrtf(var + TU_EPS);
  }
}

__global__ void tu_inapply_kernel(const float* __restrict__ x, const float* __restrict__ stats,
                                  const float* __restrict__ g, const float* __restrict__ b,
                                  float* __restrict__ y, int C, int n){
  int i = blockIdx.x*blockDim.x + threadIdx.x; if(i>=n) return;
  int c = i % C;
  float v = (x[i]-stats[2*c])*stats[2*c+1]*g[c] + b[c];
  y[i] = tu_elu(v);
}

// ------------------------------------------------------------ generic GEMM
// out[M_TOK,N] = act( A[M_TOK,K] * W + bias ).  Block: 128 rows x 128 cols.
// Weight K-chunk staged in LDS (async b128 where chunks are fully valid),
// each wave: 1 A-load -> 8 WMMAs across column tiles.
template<int ACT>
__global__ void tu_gemm_kernel(const float* __restrict__ A, const float* __restrict__ W,
                               const float* __restrict__ bias, float* __restrict__ out,
                               int N, int K, int ldw, int woff, int wt_nk){
  __shared__ float lw[32*128];                 // 16 KB (KN: [k][c], NK: [c][k])
  int tid = threadIdx.x;
  int wave = tid >> 5, lane = tid & 31;
  int ibase = blockIdx.x*128 + wave*16;
  int cb    = blockIdx.y*128;
  int mloc = lane & 15;
  int kb   = (lane >> 4) << 3;
  int koff = (lane >> 4) << 4;
  int top  = (lane >= 16) ? 8 : 0;
  const float* arow = A + (size_t)(ibase + mloc) * K;
  v8f zero = {0.f,0.f,0.f,0.f,0.f,0.f,0.f,0.f};
  v8f acc[8];
  #pragma unroll
  for(int t=0;t<8;t++) acc[t] = zero;

  for(int kk=0; kk<K; kk+=32){
    // ---- stage weight slab [kk..kk+31] x [cb..cb+127] into LDS
    if(wt_nk){
      #pragma unroll
      for(int q=0;q<4;q++){
        int id = tid + 256*q;                  // 0..1023
        int c  = id >> 3;                      // 0..127
        int k4 = (id & 7) << 2;                // 0,4,..,28
        int gn = cb + c;
        unsigned loff = (unsigned)(size_t)&lw[c*32 + k4];
        if(gn < N && kk + k4 + 3 < K){
          tu_async_b128(loff, W + (size_t)(woff+gn)*ldw + kk + k4);
        } else {
          for(int e=0;e<4;e++){
            int k = kk + k4 + e;
            lw[c*32 + k4 + e] = (gn<N && k<K) ? W[(size_t)(woff+gn)*ldw + k] : 0.f;
          }
        }
      }
    } else {
      #pragma unroll
      for(int q=0;q<4;q++){
        int id = tid + 256*q;
        int k  = id >> 5;                      // 0..31
        int c4 = (id & 31) << 2;               // 0,4,..,124
        int gk = kk + k;
        int gn = cb + c4;
        unsigned loff = (unsigned)(size_t)&lw[k*128 + c4];
        if(gk < K && gn + 3 < N){
          tu_async_b128(loff, W + (size_t)gk*ldw + woff + gn);
        } else {
          for(int e=0;e<4;e++)
            lw[k*128 + c4 + e] = (gk<K && gn+e<N) ? W[(size_t)gk*ldw + woff + gn + e] : 0.f;
        }
      }
    }
    tu_wait_async();
    __syncthreads();

    if(kk + 32 < K) __builtin_prefetch(arow + kk + 32, 0, 0);

    v16h a;
    #pragma unroll
    for(int e=0;e<8;e++){
      int k1 = kk + koff + e, k2 = kk + koff + 8 + e;
      a[e]   = (k1<K) ? (_Float16)arow[k1] : (_Float16)0.f;
      a[8+e] = (k2<K) ? (_Float16)arow[k2] : (_Float16)0.f;
    }
    #pragma unroll
    for(int nt=0;nt<8;nt++){
      int cn = nt*16 + mloc;
      v16h bv;
      if(wt_nk){
        #pragma unroll
        for(int e=0;e<16;e++) bv[e] = (_Float16)lw[cn*32 + koff + e];
      } else {
        #pragma unroll
        for(int e=0;e<16;e++) bv[e] = (_Float16)lw[(koff+e)*128 + cn];
      }
      acc[nt] = __builtin_amdgcn_wmma_f32_16x16x32_f16(false, a, false, bv, (short)0, acc[nt], false, false);
    }
    __syncthreads();
  }
  #pragma unroll
  for(int nt=0;nt<8;nt++){
    int n = cb + nt*16 + mloc;
    if(n < N){
      float bz = bias ? bias[n] : 0.f;
      #pragma unroll
      for(int r=0;r<8;r++){
        float v = acc[nt][r] + bz;
        if(ACT==1) v = v>0.f ? v : 0.f;
        if(ACT==2) v = tu_sig(v);
        out[(size_t)(ibase + r + top) * N + n] = v;
      }
    }
  }
}

// -------------------------------------------- triangle einsum (batched GEMM)
// AT,BT: [64][256][256] (k contiguous).  Block: 128 i x 128 j for one d.
// BT slab async-staged in LDS; 8 WMMAs per A-load.
__global__ void tu_trimul_kernel(const float* __restrict__ AT, const float* __restrict__ BT,
                                 float* __restrict__ prod){
  __shared__ float lb[128*32];                 // [j][k] f32, 16 KB
  int tid = threadIdx.x, wave = tid >> 5, lane = tid & 31;
  int d  = blockIdx.y;
  int rs = blockIdx.x & 1, cs = blockIdx.x >> 1;
  int ibase = rs*128 + wave*16;
  int jb = cs*128;
  int mloc = lane & 15;
  int kb = (lane >> 4) << 3, koff = (lane >> 4) << 4;
  int top = (lane >= 16) ? 8 : 0;
  const float* Ab = AT + (size_t)d*M_TOK + (size_t)(ibase + mloc)*256;
  const float* Bb = BT + (size_t)d*M_TOK;
  v8f zero = {0.f,0.f,0.f,0.f,0.f,0.f,0.f,0.f};
  v8f acc[8];
  #pragma unroll
  for(int t=0;t<8;t++) acc[t] = zero;

  for(int kk=0; kk<256; kk+=32){
    #pragma unroll
    for(int q=0;q<4;q++){
      int id = tid + 256*q;                    // 0..1023
      int j  = id >> 3;                        // 0..127
      int k4 = (id & 7) << 2;                  // 0,4,..,28
      unsigned loff = (unsigned)(size_t)&lb[j*32 + k4];
      tu_async_b128(loff, Bb + (size_t)(jb + j)*256 + kk + k4);
    }
    tu_wait_async();
    __syncthreads();

    if(kk + 32 < 256) __builtin_prefetch(Ab + kk + 32, 0, 0);

    v16h a;
    #pragma unroll
    for(int e=0;e<8;e++){
      a[e]   = (_Float16)Ab[kk + koff + e];
      a[8+e] = (_Float16)Ab[kk + koff + 8 + e];
    }
    #pragma unroll
    for(int nt=0;nt<8;nt++){
      v16h bv;
      #pragma unroll
      for(int e=0;e<16;e++) bv[e] = (_Float16)lb[(nt*16 + mloc)*32 + koff + e];
      acc[nt] = __builtin_amdgcn_wmma_f32_16x16x32_f16(false, a, false, bv, (short)0, acc[nt], false, false);
    }
    __syncthreads();
  }
  #pragma unroll
  for(int nt=0;nt<8;nt++)
    #pragma unroll
    for(int r=0;r<8;r++)
      prod[((size_t)(ibase + r + top)*256 + jb + nt*16 + mloc)*64 + d] = acc[nt][r];
}

// transpose ab[M,128] (a = [:,0:64], b = [:,64:128]) into [d][row][k] layouts
__global__ void tu_makeabt_kernel(const float* __restrict__ ab, float* __restrict__ AT,
                                  float* __restrict__ BT, int incoming){
  int gid = blockIdx.x*blockDim.x + threadIdx.x;
  if(gid >= 64*M_TOK) return;
  int d = gid >> 16;
  int rc = gid & 65535;
  int r = rc >> 8, c = rc & 255;
  size_t pos = incoming ? ((size_t)c*256 + r) : ((size_t)r*256 + c);
  AT[gid] = ab[pos*128 + d];
  BT[gid] = ab[pos*128 + 64 + d];
}

// -------------------------------------------------- fused triangle attention
__global__ void tu_attn_kernel(const float* __restrict__ Q, const float* __restrict__ K,
                               const float* __restrict__ V, const float* __restrict__ bias4,
                               float* __restrict__ out, int colmode){
  __shared__ _Float16 probs[4][16][256];
  int wave = threadIdx.x >> 5, lane = threadIdx.x & 31;
  int gw = blockIdx.x * 4 + wave;
  int it = gw & 15, h = (gw >> 4) & 3, f = gw >> 6;
  int ibase = it << 4;
  size_t posBase   = colmode ? (size_t)f : (size_t)f * 256;
  size_t posStride = colmode ? 256 : 1;
  int mloc = lane & 15, kb = (lane >> 4) << 3, koff = (lane >> 4) << 4;
  int top = (lane >= 16) ? 8 : 0;

  const float* qrow = Q + (posBase + (size_t)(ibase + mloc) * posStride) * 128 + h * 32;
  v16h qa;
  #pragma unroll
  for(int e=0;e<8;e++){ qa[e] = (_Float16)qrow[koff+e]; qa[8+e] = (_Float16)qrow[koff+8+e]; }

  v8f acc[16];
  for(int t=0;t<16;t++){
    const float* krow = K + (posBase + (size_t)(t*16 + mloc) * posStride) * 128 + h * 32;
    v16h kv;
    #pragma unroll
    for(int e=0;e<16;e++) kv[e] = (_Float16)krow[koff + e];
    v8f c = {0.f,0.f,0.f,0.f,0.f,0.f,0.f,0.f};
    c = __builtin_amdgcn_wmma_f32_16x16x32_f16(false, qa, false, kv, (short)0, c, false, false);
    #pragma unroll
    for(int r=0;r<8;r++){
      int qidx = ibase + r + top;
      int kidx = t*16 + mloc;
      c[r] = c[r] * 0.17677669529663687f + bias4[((size_t)qidx*256 + kidx)*4 + h];
    }
    acc[t] = c;
  }

  float rinv[8], rmax[8];
  #pragma unroll
  for(int r=0;r<8;r++){
    float mx = -1e30f;
    for(int t=0;t<16;t++) mx = fmaxf(mx, acc[t][r]);
    #pragma unroll
    for(int off=8;off>0;off>>=1) mx = fmaxf(mx, __shfl_xor(mx, off, 16));
    rmax[r] = mx;
  }
  #pragma unroll
  for(int r=0;r<8;r++){
    float s = 0.f;
    for(int t=0;t<16;t++){ float e = __expf(acc[t][r]-rmax[r]); acc[t][r] = e; s += e; }
    #pragma unroll
    for(int off=8;off>0;off>>=1) s += __shfl_xor(s, off, 16);
    rinv[r] = 1.f / s;
  }
  for(int t=0;t<16;t++)
    #pragma unroll
    for(int r=0;r<8;r++)
      probs[wave][r + top][t*16 + mloc] = (_Float16)(acc[t][r] * rinv[r]);
  __syncthreads();

  v8f o0 = {0.f,0.f,0.f,0.f,0.f,0.f,0.f,0.f};
  v8f o1 = {0.f,0.f,0.f,0.f,0.f,0.f,0.f,0.f};
  for(int js=0; js<8; js++){
    v16h pa;
    #pragma unroll
    for(int e=0;e<8;e++){
      pa[e]   = probs[wave][mloc][js*32 + koff + e];
      pa[8+e] = probs[wave][mloc][js*32 + koff + 8 + e];
    }
    v16h bv0, bv1;
    #pragma unroll
    for(int e=0;e<16;e++){
      const float* vrow = V + (posBase + (size_t)(js*32 + koff + e) * posStride) * 128 + h * 32;
      bv0[e] = (_Float16)vrow[mloc];
      bv1[e] = (_Float16)vrow[16 + mloc];
    }
    o0 = __builtin_amdgcn_wmma_f32_16x16x32_f16(false, pa, false, bv0, (short)0, o0, false, false);
    o1 = __builtin_amdgcn_wmma_f32_16x16x32_f16(false, pa, false, bv1, (short)0, o1, false, false);
  }
  #pragma unroll
  for(int r=0;r<8;r++){
    size_t qpos = posBase + (size_t)(ibase + r + top) * posStride;
    out[qpos*128 + h*32 + mloc]      = o0[r];
    out[qpos*128 + h*32 + 16 + mloc] = o1[r];
  }
}

// pair bias: bias4[m,h] = zn[m,:] @ pw[:,h]
__global__ void tu_pairbias_kernel(const float* __restrict__ zn, const float* __restrict__ pw,
                                   float* __restrict__ bias4){
  int i = blockIdx.x*blockDim.x + threadIdx.x;
  if(i >= M_TOK*4) return;
  int m = i >> 2, h = i & 3;
  const float* zr = zn + (size_t)m*128;
  float s = 0.f;
  for(int k=0;k<128;k++) s += zr[k]*pw[k*4+h];
  bias4[i] = s;
}

// ------------------------------------------- 3x3 conv (implicit im2col WMMA)
__global__ void tu_conv3_kernel(const float* __restrict__ in, const float* __restrict__ w,
                                const float* __restrict__ bias, float* __restrict__ out){
  int wave = threadIdx.x >> 5, lane = threadIdx.x & 31;
  int tile = blockIdx.x * 8 + wave;
  int tn = tile & 3, tm = tile >> 2;
  int pbase = tm << 4, nbase = tn << 4;
  int mloc = lane & 15, kb = (lane >> 4) << 3, koff = (lane >> 4) << 4;
  int top = (lane >= 16) ? 8 : 0;
  int p = pbase + mloc;
  int pi = p >> 8, pj = p & 255;
  int co = nbase + mloc;
  v8f acc = {0.f,0.f,0.f,0.f,0.f,0.f,0.f,0.f};
  for(int dy=0; dy<3; dy++)
  for(int dx=0; dx<3; dx++){
    int ni = pi + dy - 1, nj = pj + dx - 1;
    bool valid = ((unsigned)ni < 256u) && ((unsigned)nj < 256u);
    const float* irow = in + ((size_t)(valid ? ni*256 + nj : 0)) * 52;
    const float* wtap = w + dy*3 + dx;
    for(int kk=0; kk<52; kk+=32){
      v16h a, bv;
      #pragma unroll
      for(int e=0;e<8;e++){
        int c1 = kk + koff + e, c2 = kk + koff + 8 + e;
        a[e]   = (valid && c1<52) ? (_Float16)irow[c1] : (_Float16)0.f;
        a[8+e] = (valid && c2<52) ? (_Float16)irow[c2] : (_Float16)0.f;
      }
      #pragma unroll
      for(int e=0;e<16;e++){
        int ci = kk + koff + e;
        bv[e] = (ci<52 && co<52) ? (_Float16)wtap[((size_t)co*52 + ci)*9] : (_Float16)0.f;
      }
      acc = __builtin_amdgcn_wmma_f32_16x16x32_f16(false, a, false, bv, (short)0, acc, false, false);
    }
  }
  if(co < 52){
    float bz = bias[co];
    #pragma unroll
    for(int r=0;r<8;r++) out[(size_t)(pbase + r + top)*52 + co] = acc[r] + bz;
  }
}

// ----------------------------------------------------------- elementwise ops
__global__ void tu_mul_kernel(float* __restrict__ d, const float* __restrict__ a,
                              const float* __restrict__ b, int n){
  int i = blockIdx.x*blockDim.x + threadIdx.x; if(i<n) d[i] = a[i]*b[i];
}
__global__ void tu_gadd3_kernel(float* __restrict__ d, const float* __restrict__ z,
                                const float* __restrict__ g, const float* __restrict__ p,
                                const float* __restrict__ cv, int n){
  int i = blockIdx.x*blockDim.x + threadIdx.x; if(i<n) d[i] = z[i] + g[i]*p[i] + cv[i];
}
__global__ void tu_add3_kernel(float* __restrict__ d, const float* __restrict__ z,
                               const float* __restrict__ a, const float* __restrict__ b, int n){
  int i = blockIdx.x*blockDim.x + threadIdx.x; if(i<n) d[i] = z[i] + a[i] + b[i];
}
__global__ void tu_add2_kernel(float* __restrict__ d, const float* __restrict__ a,
                               const float* __restrict__ b, int n){
  int i = blockIdx.x*blockDim.x + threadIdx.x; if(i<n) d[i] = a[i] + b[i];
}
__global__ void tu_slice_kernel(float* __restrict__ dst, int ds, int doff,
                                const float* __restrict__ a, int as, int aoff,
                                const float* __restrict__ b, int bs, int boff,
                                int C, int n){
  int i = blockIdx.x*blockDim.x + threadIdx.x; if(i>=n) return;
  int m = i / C, c = i % C;
  float v = a[(size_t)m*as + aoff + c];
  if(b) v += b[(size_t)m*bs + boff + c];
  dst[(size_t)m*ds + doff + c] = v;
}

// ---------------------------------------------------------------- host side
static inline int tu_cdiv(int a, int b){ return (a + b - 1) / b; }

extern "C" void kernel_launch(void* const* d_in, const int* in_sizes, int n_in,
                              void* d_out, int out_size, void* d_ws, size_t ws_size,
                              hipStream_t stream){
  (void)in_sizes; (void)n_in; (void)out_size; (void)ws_size;
  int ix = 0;
  auto F = [&](int i){ return (const float*)d_in[i]; };
  const float* Zin = F(ix++);

  struct TA { const float *g_b,*g_w,*norm_b,*norm_g,*o_b,*o_w,*pair_w,*qkv_w; };
  struct CB { const float *bn1_b,*bn1_g,*bn3_b,*bn3_g,*bns_b,*bns_g,*c1_b,*c1_w,*c3_b,*c3_w,*cv_b,*cv_w; };
  struct TM { const float *g_b,*g_w,*l1_b,*l1_w,*l2_b,*l2_w,*norm_b,*norm_g,*o_b,*o_w,*on_b,*on_g; };
  struct TR { const float *l1_b,*l1_w,*l2_b,*l2_w,*norm_b,*norm_g; };
  auto loadTA=[&](TA& t){ t.g_b=F(ix++); t.g_w=F(ix++); t.norm_b=F(ix++); t.norm_g=F(ix++);
                          t.o_b=F(ix++); t.o_w=F(ix++); t.pair_w=F(ix++); t.qkv_w=F(ix++); };
  auto loadCB=[&](CB& c){ c.bn1_b=F(ix++); c.bn1_g=F(ix++); c.bn3_b=F(ix++); c.bn3_g=F(ix++);
                          c.bns_b=F(ix++); c.bns_g=F(ix++); c.c1_b=F(ix++); c.c1_w=F(ix++);
                          c.c3_b=F(ix++); c.c3_w=F(ix++); c.cv_b=F(ix++); c.cv_w=F(ix++); };
  auto loadTM=[&](TM& t){ t.g_b=F(ix++); t.g_w=F(ix++); t.l1_b=F(ix++); t.l1_w=F(ix++);
                          t.l2_b=F(ix++); t.l2_w=F(ix++); t.norm_b=F(ix++); t.norm_g=F(ix++);
                          t.o_b=F(ix++); t.o_w=F(ix++); t.on_b=F(ix++); t.on_g=F(ix++); };
  auto loadTR=[&](TR& t){ t.l1_b=F(ix++); t.l1_w=F(ix++); t.l2_b=F(ix++); t.l2_w=F(ix++);
                          t.norm_b=F(ix++); t.norm_g=F(ix++); };
  TA ca; loadTA(ca);
  CB cs[4]; for(int i=0;i<4;i++) loadCB(cs[i]);
  TM mi; loadTM(mi);
  TM mo; loadTM(mo);
  TA ra; loadTA(ra);
  TR tr; loadTR(tr);

  float* W   = (float*)d_ws;
  const size_t SL = (size_t)M_TOK * 128;
  float* ZCUR = W;
  float* ZN   = W + 1*SL;
  float* T1   = W + 2*SL;
  float* T2   = W + 3*SL;
  float* GATE = W + 4*SL;
  float* TMP  = W + 5*SL;
  float* CONV = W + 6*SL;
  float* POOL = W + 7*SL;
  float* AT    = POOL;
  float* BT    = AT    + (size_t)64*M_TOK;
  float* PROD  = BT    + (size_t)64*M_TOK;
  float* PRODN = PROD  + (size_t)64*M_TOK;
  float* C208  = PRODN + (size_t)64*M_TOK;
  float* O208  = C208  + (size_t)M_TOK*208;
  float* SP    = O208  + (size_t)M_TOK*208;
  float* SPN   = SP    + (size_t)M_TOK*52;
  float* PIECE = SPN   + (size_t)M_TOK*52;
  float* BIAS4 = PIECE + (size_t)M_TOK*52;
  float* STATS = BIAS4 + (size_t)M_TOK*4;
  float* VBUF  = AT;
  float* H512  = AT;
  float* OUT   = (float*)d_out;

  const int NE = M_TOK*128;
  auto LN = [&](const float* x, const float* g, const float* b, float* y, int C){
    tu_ln_kernel<<<M_TOK/8, 256, 0, stream>>>(x, g, b, y, C);
  };
  auto GEMM = [&](int act, const float* A, const float* Wt, const float* bias, float* outp,
                  int N, int K, int ldw, int woff, int nk){
    dim3 grid(M_TOK/128, tu_cdiv(N,128));
    if(act==0)      tu_gemm_kernel<0><<<grid,256,0,stream>>>(A,Wt,bias,outp,N,K,ldw,woff,nk);
    else if(act==1) tu_gemm_kernel<1><<<grid,256,0,stream>>>(A,Wt,bias,outp,N,K,ldw,woff,nk);
    else            tu_gemm_kernel<2><<<grid,256,0,stream>>>(A,Wt,bias,outp,N,K,ldw,woff,nk);
  };
  auto TRIMUL = [&](const TM& p, int incoming, const float* zin){
    LN(zin, p.norm_g, p.norm_b, ZN, 128);
    GEMM(0, ZN, p.l1_w, p.l1_b, T1, 128, 128, 128, 0, 0);
    GEMM(2, ZN, p.l2_w, p.l2_b, T2, 128, 128, 128, 0, 0);
    tu_mul_kernel<<<tu_cdiv(NE,256),256,0,stream>>>(T2, T2, T1, NE);
    GEMM(2, ZN, p.g_w, p.g_b, GATE, 128, 128, 128, 0, 0);
    tu_makeabt_kernel<<<tu_cdiv(64*M_TOK,256),256,0,stream>>>(T2, AT, BT, incoming);
    tu_trimul_kernel<<<dim3(4,64),256,0,stream>>>(AT, BT, PROD);
    LN(PROD, p.on_g, p.on_b, PRODN, 64);
    GEMM(0, PRODN, p.o_w, p.o_b, TMP, 128, 64, 128, 0, 0);
  };
  auto CONVST = [&](const CB& p, const float* zin){
    tu_instat_kernel<<<128,256,0,stream>>>(zin, 128, STATS);
    tu_inapply_kernel<<<tu_cdiv(NE,256),256,0,stream>>>(zin, STATS, p.bn1_g, p.bn1_b, ZN, 128, NE);
    GEMM(0, ZN, p.c1_w, p.c1_b, C208, 208, 128, 128, 0, 1);
    const int N52 = M_TOK*52;
    for(int i=0;i<3;i++){
      if(i==0) tu_slice_kernel<<<tu_cdiv(N52,256),256,0,stream>>>(SP,52,0, C208,208,0,    nullptr,0,0, 52,N52);
      else     tu_slice_kernel<<<tu_cdiv(N52,256),256,0,stream>>>(SP,52,0, C208,208,i*52, PIECE,52,0,  52,N52);
      tu_instat_kernel<<<52,256,0,stream>>>(SP, 52, STATS);
      tu_inapply_kernel<<<tu_cdiv(N52,256),256,0,stream>>>(SP, STATS, p.bns_g+i*52, p.bns_b+i*52, SPN, 52, N52);
      tu_conv3_kernel<<<2048,256,0,stream>>>(SPN, p.cv_w + (size_t)i*52*52*9, p.cv_b + i*52, PIECE);
      tu_slice_kernel<<<tu_cdiv(N52,256),256,0,stream>>>(O208,208,i*52, PIECE,52,0, nullptr,0,0, 52,N52);
    }
    tu_slice_kernel<<<tu_cdiv(N52,256),256,0,stream>>>(O208,208,156, C208,208,156, nullptr,0,0, 52,N52);
    tu_instat_kernel<<<208,256,0,stream>>>(O208, 208, STATS);
    tu_inapply_kernel<<<tu_cdiv(M_TOK*208,256),256,0,stream>>>(O208, STATS, p.bn3_g, p.bn3_b, C208, 208, M_TOK*208);
    GEMM(0, C208, p.c3_w, p.c3_b, CONV, 128, 208, 208, 0, 1);
  };
  auto ATTN = [&](const TA& p, int colmode, const float* zin){
    LN(zin, p.norm_g, p.norm_b, ZN, 128);
    GEMM(0, ZN, p.qkv_w, nullptr, T1,   128, 128, 384, 0,   0);
    GEMM(0, ZN, p.qkv_w, nullptr, T2,   128, 128, 384, 128, 0);
    GEMM(0, ZN, p.qkv_w, nullptr, VBUF, 128, 128, 384, 256, 0);
    tu_pairbias_kernel<<<tu_cdiv(M_TOK*4,256),256,0,stream>>>(ZN, p.pair_w, BIAS4);
    GEMM(2, ZN, p.g_w, p.g_b, GATE, 128, 128, 128, 0, 0);
    tu_attn_kernel<<<4096,128,0,stream>>>(T1, T2, VBUF, BIAS4, TMP, colmode);
    tu_mul_kernel<<<tu_cdiv(NE,256),256,0,stream>>>(T1, GATE, TMP, NE);
    GEMM(0, T1, p.o_w, p.o_b, T2, 128, 128, 128, 0, 0);
  };

  TRIMUL(mo, 0, Zin);  CONVST(cs[0], Zin);
  tu_gadd3_kernel<<<tu_cdiv(NE,256),256,0,stream>>>(ZCUR, Zin, GATE, TMP, CONV, NE);
  TRIMUL(mi, 1, ZCUR); CONVST(cs[1], ZCUR);
  tu_gadd3_kernel<<<tu_cdiv(NE,256),256,0,stream>>>(OUT, ZCUR, GATE, TMP, CONV, NE);
  ATTN(ra, 0, OUT);    CONVST(cs[2], OUT);
  tu_add3_kernel<<<tu_cdiv(NE,256),256,0,stream>>>(ZCUR, OUT, T2, CONV, NE);
  ATTN(ca, 1, ZCUR);   CONVST(cs[3], ZCUR);
  tu_add3_kernel<<<tu_cdiv(NE,256),256,0,stream>>>(OUT, ZCUR, T2, CONV, NE);
  LN(OUT, tr.norm_g, tr.norm_b, ZN, 128);
  GEMM(1, ZN,   tr.l1_w, tr.l1_b, H512, 512, 128, 512, 0, 0);
  GEMM(0, H512, tr.l2_w, tr.l2_b, TMP,  128, 512, 128, 0, 0);
  tu_add2_kernel<<<tu_cdiv(NE,256),256,0,stream>>>(OUT, OUT, TMP, NE);
}
